// Sparsity_Checker_56745107915393
// MI455X (gfx1250) — compile-verified, matrix operand order checked by reading, not run on hardware
//
#include <hip/hip_runtime.h>

// CDNA5 / gfx1250, wave32.
// Fused streaming kernel: copy spikes -> out (non-temporal b128, 411MB stream,
// no reuse -> don't thrash the 192MB L2), histogram collector = sum_t spikes[t].
// Finalize kernel: column-sum of per-block f16 partial histograms via
// v_wmma_f32_16x16x32_f16 with A = ones (D[m][n] = sum_k B[k][n]).

typedef __attribute__((ext_vector_type(16))) _Float16 v16h;
typedef __attribute__((ext_vector_type(8)))  float    v8f;
typedef __attribute__((ext_vector_type(4)))  float    v4f;

#define BINS_PAD 16   // padded histogram width (T+1 <= 16), 16 halves = 32B rows

__global__ __launch_bounds__(256)
void snn_stream_hist(const float* __restrict__ spikes,
                     float* __restrict__ out,
                     _Float16* __restrict__ partials,   // column-major [16][Ppad]
                     long long total, int T, int P, long long Ppad)
{
    __shared__ unsigned int lbins[BINS_PAD];
    const int tid  = threadIdx.x;
    const int lane = tid & 31;

    // Block 0 zeroes the pad rows [P, Ppad) of every column so the WMMA
    // finalize loop can run unguarded over Ppad rows.
    if (blockIdx.x == 0) {
        long long padCount = (Ppad - (long long)P) * BINS_PAD;
        for (long long i = tid; i < padCount; i += blockDim.x) {
            long long r   = (long long)P + (i >> 4);
            long long col = i & 15;
            partials[col * Ppad + r] = (_Float16)0.0f;
        }
    }

    if (tid < BINS_PAD) lbins[tid] = 0u;
    __syncthreads();

    const bool vec4ok = ((total & 3LL) == 0LL);
    const bool packedPath = (T <= 4);   // 12-bit fields x 5 bins in u64

    unsigned long long packed = 0ULL;

    const long long nQuads  = (total + 3) >> 2;
    const long long strideQ = (long long)gridDim.x * blockDim.x;
    for (long long q = (long long)blockIdx.x * blockDim.x + tid; q < nQuads; q += strideQ) {
        const long long base = q << 2;
        if (vec4ok && T == 4 && base + 4 <= total) {
            // Hot path (T==4): 4 clustered NT b128 loads, 4 NT b128 stores.
            const v4f* p0 = reinterpret_cast<const v4f*>(spikes + base);
            const v4f* p1 = reinterpret_cast<const v4f*>(spikes + total + base);
            const v4f* p2 = reinterpret_cast<const v4f*>(spikes + 2 * total + base);
            const v4f* p3 = reinterpret_cast<const v4f*>(spikes + 3 * total + base);
            v4f v0 = __builtin_nontemporal_load(p0);
            v4f v1 = __builtin_nontemporal_load(p1);
            v4f v2 = __builtin_nontemporal_load(p2);
            v4f v3 = __builtin_nontemporal_load(p3);
            __builtin_nontemporal_store(v0, reinterpret_cast<v4f*>(out + base));
            __builtin_nontemporal_store(v1, reinterpret_cast<v4f*>(out + total + base));
            __builtin_nontemporal_store(v2, reinterpret_cast<v4f*>(out + 2 * total + base));
            __builtin_nontemporal_store(v3, reinterpret_cast<v4f*>(out + 3 * total + base));
            v4f acc = v0 + v1 + v2 + v3;
            float a4[4] = {acc.x, acc.y, acc.z, acc.w};
            #pragma unroll
            for (int e = 0; e < 4; ++e) {
                int c = (int)(a4[e] + 0.5f);
                c = c < 0 ? 0 : (c > 4 ? 4 : c);
                packed += 1ULL << (12 * c);
            }
        } else if (vec4ok && base + 4 <= total) {
            v4f acc = {0.f, 0.f, 0.f, 0.f};
            for (int t = 0; t < T; ++t) {
                const long long off = (long long)t * total + base;
                v4f v = __builtin_nontemporal_load(
                            reinterpret_cast<const v4f*>(spikes + off));
                __builtin_nontemporal_store(v, reinterpret_cast<v4f*>(out + off));
                acc += v;
            }
            float a4[4] = {acc.x, acc.y, acc.z, acc.w};
            #pragma unroll
            for (int e = 0; e < 4; ++e) {
                int c = (int)(a4[e] + 0.5f);
                c = c < 0 ? 0 : (c > T ? T : c);
                if (packedPath) packed += 1ULL << (12 * c);
                else            atomicAdd(&lbins[c], 1u);
            }
        } else {
            for (long long i = base; i < total && i < base + 4; ++i) {
                float s = 0.f;
                for (int t = 0; t < T; ++t) {
                    const long long off = (long long)t * total + i;
                    float v = spikes[off];
                    out[off] = v;
                    s += v;
                }
                int c = (int)(s + 0.5f);
                c = c < 0 ? 0 : (c > T ? T : c);
                if (packedPath) packed += 1ULL << (12 * c);
                else            atomicAdd(&lbins[c], 1u);
            }
        }
    }

    if (packedPath) {
        // wave32 butterfly reduction of the packed 5x12-bit counters
        for (int off = 16; off > 0; off >>= 1)
            packed += __shfl_down(packed, off, 32);
        if (lane == 0) {
            for (int b = 0; b <= T; ++b) {
                unsigned int v = (unsigned int)((packed >> (12 * b)) & 0xFFFULL);
                if (v) atomicAdd(&lbins[b], v);
            }
        }
    }
    __syncthreads();

    // Per-block partial histogram as f16 (exact: counts/block <= 1024 < 2048)
    if (tid < BINS_PAD) {
        unsigned int v = (tid <= T) ? lbins[tid] : 0u;
        partials[(long long)tid * Ppad + blockIdx.x] = (_Float16)(float)v;
    }
}

// Single-wave finalize: column-sums of partials via WMMA with A = ones.
// B tile = 32 rows x 16 cols of f16; lane's column n = lane & 15; lanes 0-15
// carry rows r0..r0+15, lanes 16-31 carry rows r0+16..r0+31. With A all-ones,
// D[m][n] = sum_k B[k][n] regardless of the exact K<->slot mapping, so the
// only layout fact used is the documented N = lane%16 of C/D (VGPR0, M=0).
__global__ void snn_finalize(const _Float16* __restrict__ partials,  // [16][Ppad]
                             long long Ppad,
                             float* __restrict__ outTail,
                             long long total, int T)
{
    const int lane    = threadIdx.x;       // 32 threads, EXEC all ones
    const int n       = lane & 15;
    const int halfSel = lane >> 4;

    v16h a;
    #pragma unroll
    for (int h = 0; h < 16; ++h) a[h] = (_Float16)1.0f;

    v8f c = {};
    const _Float16* colp = partials + (long long)n * Ppad;

    for (long long r0 = 0; r0 < Ppad; r0 += 32) {
        const long long rbase = r0 + (long long)halfSel * 16;
        v16h b = *reinterpret_cast<const v16h*>(colp + rbase);  // 16 contiguous rows
        c = __builtin_amdgcn_wmma_f32_16x16x32_f16(
                /*neg_a=*/false, a, /*neg_b=*/false, b,
                /*c_mod=*/(short)0, c, /*reuse_a=*/false, /*reuse_b=*/false);
    }

    __shared__ float bins[16];
    if (lane < 16) bins[lane] = c[0];   // D: VGPR0, lanes 0-15 -> M=0, N=lane
    __syncthreads();

    if (lane == 0) {
        double wsum = 0.0;
        for (int k = 0; k <= T; ++k) wsum += (double)k * (double)bins[k];
        const double tt = (double)total * (double)T;
        outTail[0] = (float)((tt - wsum) / tt);                  // sparsity_ratio
        outTail[1] = (float)((double)bins[0] / (double)total);   // coll_zero_frac
        for (int k = 0; k <= T; ++k) outTail[2 + k] = bins[k];   // unique_counts
    }
}

extern "C" void kernel_launch(void* const* d_in, const int* in_sizes, int n_in,
                              void* d_out, int out_size, void* d_ws, size_t ws_size,
                              hipStream_t stream)
{
    const float* spikes = (const float*)d_in[0];
    const long long nSpk = (long long)in_sizes[0];

    // out = [spikes(nSpk), sparsity_ratio(1), coll_zero_frac(1), bins(T+1)]
    int T = out_size - in_sizes[0] - 3;
    if (T < 1)  T = 1;
    if (T > 15) T = 15;
    const long long total = nSpk / T;

    float*     out      = (float*)d_out;
    _Float16*  partials = (_Float16*)d_ws;

    const int block = 256;
    const long long elemsPerBlock = (long long)block * 4;   // 1024 elems/block
    long long nBlocks = (total + elemsPerBlock - 1) / elemsPerBlock;
    if (nBlocks < 1) nBlocks = 1;

    long long Ppad = ((nBlocks + 31) / 32) * 32;
    long long PpadMax = (long long)(ws_size / (BINS_PAD * sizeof(_Float16)));
    PpadMax = (PpadMax / 32) * 32;
    if (PpadMax < 32) PpadMax = 32;
    if (Ppad > PpadMax) { Ppad = PpadMax; nBlocks = Ppad; }  // grid-stride covers rest

    snn_stream_hist<<<dim3((unsigned)nBlocks), dim3(block), 0, stream>>>(
        spikes, out, partials, total, T, (int)nBlocks, Ppad);

    snn_finalize<<<dim3(1), dim3(32), 0, stream>>>(
        partials, Ppad, out + nSpk, total, T);
}
